// MultiHeadAttention_5634997092674
// MI455X (gfx1250) — matrix operand order repaired and checked by reading, not compiled, not run on hardware
//
#include <hip/hip_runtime.h>

#define S_LEN   4096
#define D_MODEL 1024
#define NHEADS  16
#define HDIM    64
#define ATTN_SCALE 0.125f   // 1/sqrt(64)

typedef __attribute__((ext_vector_type(16))) _Float16 v16h;
typedef __attribute__((ext_vector_type(8)))  _Float16 v8h;
typedef __attribute__((ext_vector_type(8)))  float    v8f;
typedef __attribute__((ext_vector_type(4)))  float    v4f;

union FragU { v16h v; v8h h[2]; };

// A-matrix (16x32 f16) fragment: lanes 0-15 hold K in {kk+0..7, kk+16..23},
// lanes 16-31 hold K in {kk+8..15, kk+24..31}  (CDNA5 ISA 7.12.2)
__device__ __forceinline__ v16h ldsFragA(const _Float16* rowp, int kk, int hlf) {
  const _Float16* p = rowp + kk + hlf * 8;
  FragU u;
  u.h[0] = *(const v8h*)(p);
  u.h[1] = *(const v8h*)(p + 16);
  return u.v;
}

// B-matrix (32x16 f16) fragment: lanes 0-15 hold K in {kk+0..7, kk+16..23},
// lanes 16-31 hold K in {kk+8..15, kk+24..31} (same order as A)
__device__ __forceinline__ v16h ldsFragB(const _Float16* rowp, int kk, int hlf) {
  const _Float16* p = rowp + kk + hlf * 8;
  FragU u;
  u.h[0] = *(const v8h*)(p);
  u.h[1] = *(const v8h*)(p + 16);
  return u.v;
}

__device__ __forceinline__ v8f wmma_f16(v16h a, v16h b, v8f c) {
  return __builtin_amdgcn_wmma_f32_16x16x32_f16(false, a, false, b, (short)0, c,
                                                false, false);
}

// CDNA5 async global->LDS copy (tracked by ASYNCcnt, writes LDS directly).
// Low 32 bits of a generic pointer to __shared__ data are the DS byte address.
__device__ __forceinline__ void asyncLoadB128(_Float16* lds_dst,
                                              const _Float16* gsrc) {
  unsigned l = (unsigned)(unsigned long long)lds_dst;
  unsigned long long g = (unsigned long long)gsrc;
  asm volatile("global_load_async_to_lds_b128 %0, %1, off"
               :
               : "v"(l), "v"(g)
               : "memory");
}

__device__ __forceinline__ void waitAsync() {
  asm volatile("s_wait_asynccnt 0x0" ::: "memory");
}

// ---------------------------------------------------------------- fp32 -> fp16
__global__ void cvt_f32_f16(const float* __restrict__ in,
                            _Float16* __restrict__ out, int n) {
  int stride = gridDim.x * blockDim.x * 8;
  for (int i = (blockIdx.x * blockDim.x + threadIdx.x) * 8; i < n; i += stride) {
    v4f a = *(const v4f*)(in + i);
    v4f b = *(const v4f*)(in + i + 4);
    v8h o;
#pragma unroll
    for (int j = 0; j < 4; ++j) { o[j] = (_Float16)a[j]; o[j + 4] = (_Float16)b[j]; }
    *(v8h*)(out + i) = o;
  }
}

// -------------------------------------------------- C[M,N] = A[M,K] @ B[N,K]^T + bias
// mode 0: fp32 store to C[M][N]
// mode 1: fp16 store to C[M][N]
// mode 2: fp16 store transposed: C[N][M]   (used for V)
#define BLK_M 128
#define BLK_N 128
#define BLK_K 64
#define GLDS  72   // padded LDS row stride (halves); 144B keeps 16B align, no bank conflict

__global__ __launch_bounds__(256)
void gemm_xwT(const _Float16* __restrict__ A, const _Float16* __restrict__ B,
              const float* __restrict__ bias, void* __restrict__ Cv,
              int M, int N, int K, int mode) {
  __shared__ __align__(16) _Float16 As[2][BLK_M * GLDS];
  __shared__ __align__(16) _Float16 Bs[2][BLK_N * GLDS];

  const int tid  = threadIdx.x;
  const int wid  = tid >> 5;
  const int lane = tid & 31;
  const int hlf  = lane >> 4;
  const int rr   = lane & 15;
  const int wm   = wid & 3;   // 4 waves along M
  const int wn   = wid >> 2;  // 2 waves along N
  const int m0   = blockIdx.y * BLK_M;
  const int n0   = blockIdx.x * BLK_N;

  // issue one K-tile's async copies: 128 rows x 64 halves = 1024 16B chunks/tile
  auto issueTile = [&](int k0, int buf) {
#pragma unroll
    for (int t = 0; t < 4; ++t) {
      int c   = tid + t * 256;
      int row = c >> 3;
      int cc  = (c & 7) * 8;
      asyncLoadB128(&As[buf][row * GLDS + cc],
                    A + (size_t)(m0 + row) * K + k0 + cc);
      asyncLoadB128(&Bs[buf][row * GLDS + cc],
                    B + (size_t)(n0 + row) * K + k0 + cc);
    }
  };

  v8f acc[2][4];
#pragma unroll
  for (int mi = 0; mi < 2; ++mi)
#pragma unroll
    for (int ni = 0; ni < 4; ++ni)
      acc[mi][ni] = (v8f){0.f, 0.f, 0.f, 0.f, 0.f, 0.f, 0.f, 0.f};

  issueTile(0, 0);
  waitAsync();
  __syncthreads();

  const int nk = K / BLK_K;
  for (int t = 0; t < nk; ++t) {
    const int buf = t & 1;
    if (t + 1 < nk) issueTile((t + 1) * BLK_K, buf ^ 1);  // overlap with compute
#pragma unroll
    for (int kk = 0; kk < BLK_K; kk += 32) {
      v16h af[2], bf[4];
#pragma unroll
      for (int mi = 0; mi < 2; ++mi)
        af[mi] = ldsFragA(&As[buf][(wm * 32 + mi * 16 + rr) * GLDS], kk, hlf);
#pragma unroll
      for (int ni = 0; ni < 4; ++ni)
        bf[ni] = ldsFragB(&Bs[buf][(wn * 64 + ni * 16 + rr) * GLDS], kk, hlf);
#pragma unroll
      for (int mi = 0; mi < 2; ++mi)
#pragma unroll
        for (int ni = 0; ni < 4; ++ni)
          acc[mi][ni] = wmma_f16(af[mi], bf[ni], acc[mi][ni]);
    }
    waitAsync();      // this wave's next-tile copies landed
    __syncthreads();  // everyone done reading buf and done filling buf^1
  }

  // epilogue: C/D layout -> element (m = i + hlf*8, n = rr) per 16x16 tile
#pragma unroll
  for (int ni = 0; ni < 4; ++ni) {
    int n = n0 + wn * 64 + ni * 16 + rr;
    float bv = bias[n];
#pragma unroll
    for (int mi = 0; mi < 2; ++mi) {
      int mb = m0 + wm * 32 + mi * 16 + hlf * 8;
      if (mode == 0) {
        float* C = (float*)Cv;
#pragma unroll
        for (int i = 0; i < 8; ++i)
          C[(size_t)(mb + i) * N + n] = acc[mi][ni][i] + bv;
      } else if (mode == 1) {
        _Float16* C = (_Float16*)Cv;
#pragma unroll
        for (int i = 0; i < 8; ++i)
          C[(size_t)(mb + i) * N + n] = (_Float16)(acc[mi][ni][i] + bv);
      } else {
        _Float16* C = (_Float16*)Cv;
        v8h pk;
#pragma unroll
        for (int i = 0; i < 8; ++i) pk[i] = (_Float16)(acc[mi][ni][i] + bv);
        *(v8h*)(C + (size_t)n * M + mb) = pk;  // contiguous along m
      }
    }
  }
}

// ------------------------------------------------------------ flash attention
// Qh, Kh: fp16 [S][1024] (head h at column h*64); Vt: fp16 [1024][S] (row n = h*64+d)
// Oh out: fp16 [S][1024]
#define FQ  128
#define FK  64
#define FLD 72

__global__ __launch_bounds__(256)
void flash_attn(const _Float16* __restrict__ Qh, const _Float16* __restrict__ Kh,
                const _Float16* __restrict__ Vt, _Float16* __restrict__ Oh) {
  __shared__ __align__(16) _Float16 Qs[FQ * FLD];
  __shared__ __align__(16) _Float16 Ks[2][FK * FLD];
  __shared__ __align__(16) _Float16 Vs[2][FK * FLD];   // Vs[d][key]
  __shared__ __align__(16) _Float16 Ps[8 * 16 * FLD];  // wave-private P staging

  const int h    = blockIdx.y;
  const int q0   = blockIdx.x * FQ;
  const int tid  = threadIdx.x;
  const int wid  = tid >> 5;
  const int lane = tid & 31;
  const int hlf  = lane >> 4;
  const int rr   = lane & 15;

  // K tile (64x64) + V tile (64 dims x 64 keys): 512 chunks each, 2 per thread
  auto issueKV = [&](int k0, int buf) {
#pragma unroll
    for (int t = 0; t < 2; ++t) {
      int c   = tid + t * 256;
      int row = c >> 3;
      int cc  = (c & 7) * 8;
      asyncLoadB128(&Ks[buf][row * FLD + cc],
                    Kh + (size_t)(k0 + row) * D_MODEL + h * HDIM + cc);
      asyncLoadB128(&Vs[buf][row * FLD + cc],
                    Vt + (size_t)(h * HDIM + row) * S_LEN + k0 + cc);
    }
  };

  // Q tile: 128 rows x 64 halves = 1024 chunks, 4 per thread (async as well)
#pragma unroll
  for (int t = 0; t < 4; ++t) {
    int c   = tid + t * 256;
    int row = c >> 3;
    int cc  = (c & 7) * 8;
    asyncLoadB128(&Qs[row * FLD + cc],
                  Qh + (size_t)(q0 + row) * D_MODEL + h * HDIM + cc);
  }
  issueKV(0, 0);

  v8f o[4];
#pragma unroll
  for (int t = 0; t < 4; ++t) o[t] = (v8f){0.f, 0.f, 0.f, 0.f, 0.f, 0.f, 0.f, 0.f};
  float mr[8], lr[8];
#pragma unroll
  for (int i = 0; i < 8; ++i) { mr[i] = -1e30f; lr[i] = 0.f; }

  const int qrow0 = q0 + wid * 16 + hlf * 8;
  const int nkb   = blockIdx.x * 2 + 2;  // causal: keys up to q0+127

  waitAsync();
  __syncthreads();

  for (int kb = 0; kb < nkb; ++kb) {
    const int k0  = kb * FK;
    const int buf = kb & 1;
    if (kb + 1 < nkb) issueKV((kb + 1) * FK, buf ^ 1);  // overlap with compute

    // S = Q @ K^T  (wave handles 16 query rows x 64 keys)
    v8f s[4];
#pragma unroll
    for (int t = 0; t < 4; ++t) s[t] = (v8f){0.f, 0.f, 0.f, 0.f, 0.f, 0.f, 0.f, 0.f};
#pragma unroll
    for (int dk = 0; dk < HDIM; dk += 32) {
      v16h aq = ldsFragA(Qs + (wid * 16 + rr) * FLD, dk, hlf);
#pragma unroll
      for (int t = 0; t < 4; ++t) {
        v16h bk = ldsFragB(&Ks[buf][(t * 16 + rr) * FLD], dk, hlf);
        s[t] = wmma_f16(aq, bk, s[t]);
      }
    }

    // scale + causal mask + row max
    float rmax[8];
#pragma unroll
    for (int i = 0; i < 8; ++i) rmax[i] = -1e30f;
#pragma unroll
    for (int t = 0; t < 4; ++t) {
      int kpos = k0 + t * 16 + rr;
#pragma unroll
      for (int i = 0; i < 8; ++i) {
        float v = s[t][i] * ATTN_SCALE;
        if (kpos > qrow0 + i) v = -1e30f;
        s[t][i] = v;
        rmax[i] = fmaxf(rmax[i], v);
      }
    }
#pragma unroll
    for (int i = 0; i < 8; ++i)
#pragma unroll
      for (int off = 1; off < 16; off <<= 1)
        rmax[i] = fmaxf(rmax[i], __shfl_xor(rmax[i], off, 16));

    // online softmax update
    float corr[8], rsum[8];
#pragma unroll
    for (int i = 0; i < 8; ++i) {
      float nm = fmaxf(mr[i], rmax[i]);
      corr[i]  = __expf(mr[i] - nm);
      mr[i]    = nm;
      rsum[i]  = 0.f;
    }
#pragma unroll
    for (int t = 0; t < 4; ++t)
#pragma unroll
      for (int i = 0; i < 8; ++i) {
        float p = __expf(s[t][i] - mr[i]);
        s[t][i] = p;
        rsum[i] += p;
      }
#pragma unroll
    for (int i = 0; i < 8; ++i) {
#pragma unroll
      for (int off = 1; off < 16; off <<= 1)
        rsum[i] += __shfl_xor(rsum[i], off, 16);
      lr[i] = lr[i] * corr[i] + rsum[i];
    }
#pragma unroll
    for (int t = 0; t < 4; ++t)
#pragma unroll
      for (int i = 0; i < 8; ++i) o[t][i] *= corr[i];

    // stage P (fp16) to wave-private LDS, C-layout -> row-major
    _Float16* pb = Ps + wid * 16 * FLD;
#pragma unroll
    for (int t = 0; t < 4; ++t)
#pragma unroll
      for (int i = 0; i < 8; ++i)
        pb[(hlf * 8 + i) * FLD + t * 16 + rr] = (_Float16)s[t][i];

    // O += P @ V   (A = P [16 x 64keys], B = Vs^T chunks [keys x dim])
#pragma unroll
    for (int kk = 0; kk < FK; kk += 32) {
      v16h ap = ldsFragA(pb + rr * FLD, kk, hlf);
#pragma unroll
      for (int t = 0; t < 4; ++t) {
        v16h bv = ldsFragB(&Vs[buf][(t * 16 + rr) * FLD], kk, hlf);
        o[t] = wmma_f16(ap, bv, o[t]);
      }
    }

    waitAsync();      // next K/V tile landed
    __syncthreads();  // all waves done with buf; buf^1 ready for all
  }

  // normalize and store O (fp16, natural layout)
#pragma unroll
  for (int i = 0; i < 8; ++i) lr[i] = 1.0f / lr[i];
#pragma unroll
  for (int t = 0; t < 4; ++t)
#pragma unroll
    for (int i = 0; i < 8; ++i)
      Oh[(size_t)(qrow0 + i) * D_MODEL + h * HDIM + t * 16 + rr] =
          (_Float16)(o[t][i] * lr[i]);
}

// ---------------------------------------------------------------------- launch
extern "C" void kernel_launch(void* const* d_in, const int* in_sizes, int n_in,
                              void* d_out, int out_size, void* d_ws, size_t ws_size,
                              hipStream_t stream) {
  const float* x  = (const float*)d_in[0];
  const float* Wq = (const float*)d_in[1];
  const float* bq = (const float*)d_in[2];
  const float* Wk = (const float*)d_in[3];
  const float* bk = (const float*)d_in[4];
  const float* Wv = (const float*)d_in[5];
  const float* bv = (const float*)d_in[6];
  const float* Wo = (const float*)d_in[7];
  const float* bo = (const float*)d_in[8];
  float* out = (float*)d_out;

  const int XN = S_LEN * D_MODEL;    // 4 Mi elems
  const int WN = D_MODEL * D_MODEL;  // 1 Mi elems

  char* ws = (char*)d_ws;
  size_t off = 0;
  _Float16* xh  = (_Float16*)(ws + off); off += (size_t)XN * 2;
  _Float16* Wqh = (_Float16*)(ws + off); off += (size_t)WN * 2;
  _Float16* Wkh = (_Float16*)(ws + off); off += (size_t)WN * 2;
  _Float16* Wvh = (_Float16*)(ws + off); off += (size_t)WN * 2;
  _Float16* Woh = (_Float16*)(ws + off); off += (size_t)WN * 2;
  _Float16* Qh  = (_Float16*)(ws + off); off += (size_t)XN * 2;
  _Float16* Kh  = (_Float16*)(ws + off); off += (size_t)XN * 2;
  _Float16* Vt  = (_Float16*)(ws + off); off += (size_t)XN * 2;  // [1024][4096]
  _Float16* Oh  = (_Float16*)(ws + off); off += (size_t)XN * 2;

  cvt_f32_f16<<<XN / (256 * 8), 256, 0, stream>>>(x, xh, XN);
  cvt_f32_f16<<<WN / (256 * 8), 256, 0, stream>>>(Wq, Wqh, WN);
  cvt_f32_f16<<<WN / (256 * 8), 256, 0, stream>>>(Wk, Wkh, WN);
  cvt_f32_f16<<<WN / (256 * 8), 256, 0, stream>>>(Wv, Wvh, WN);
  cvt_f32_f16<<<WN / (256 * 8), 256, 0, stream>>>(Wo, Woh, WN);

  dim3 gg(D_MODEL / BLK_N, S_LEN / BLK_M);  // (8, 32)
  gemm_xwT<<<gg, 256, 0, stream>>>(xh, Wqh, bq, Qh, S_LEN, D_MODEL, D_MODEL, 1);
  gemm_xwT<<<gg, 256, 0, stream>>>(xh, Wkh, bk, Kh, S_LEN, D_MODEL, D_MODEL, 1);
  gemm_xwT<<<gg, 256, 0, stream>>>(xh, Wvh, bv, Vt, S_LEN, D_MODEL, D_MODEL, 2);

  flash_attn<<<dim3(S_LEN / FQ, NHEADS), 256, 0, stream>>>(Qh, Kh, Vt, Oh);

  gemm_xwT<<<gg, 256, 0, stream>>>(Oh, Woh, bo, out, S_LEN, D_MODEL, D_MODEL, 0);
}
